// NT_XentLoss_28690381537554
// MI455X (gfx1250) — matrix operand order repaired and checked by reading, not compiled, not run on hardware
//
#include <hip/hip_runtime.h>
#include <hip/hip_bf16.h>

// NT-Xent-style contrastive loss for MI455X (gfx1250, wave32).
//
// loss_i = -log(pos_i / tot_i),  pos_i = sum_{j!=i, lab_j==lab_i} sim_ij,
// tot_i = sum_j sim_ij  (temperature cancels in the ratio).
//
// sim = zn @ zn^T via v_wmma_f32_16x16x32_f16, fused with masked row
// reductions (the 8192x8192 matrix is never materialized). Column sweep is
// software-pipelined with two B-tile register buffers so B loads (L2-resident)
// overlap the WMMAs instead of serializing on s_wait_loadcnt 0.
//
// ws layout: [0, N*D*2) = zn in f16 row-major; then 512 f32 block partials.

typedef __attribute__((ext_vector_type(16))) _Float16 v16h;
typedef __attribute__((ext_vector_type(4)))  _Float16 v4h;
typedef __attribute__((ext_vector_type(8)))  float    v8f;

#define NROWS 8192
#define NDIM  128
#define NTILE (NROWS / 16)   // 512 column tiles
#define CEPS  1e-8f

union HalfPack { uint4 q[2]; v16h h; };

struct BTile { v16h b[4]; int labj; };

// ---------------------------------------------------------------------------
// Kernel 1: row-normalize z (f32 [N,128]) -> zn (f16 [N,128]).
// One wave per row; each lane owns 4 contiguous floats.
// ---------------------------------------------------------------------------
__global__ __launch_bounds__(256)
void nxt_normalize_kernel(const float* __restrict__ z, _Float16* __restrict__ zn) {
    int row  = blockIdx.x * 8 + (threadIdx.x >> 5);
    int lane = threadIdx.x & 31;
    const float4 v = *(const float4*)(z + (size_t)row * NDIM + lane * 4);
    float s = v.x * v.x + v.y * v.y + v.z * v.z + v.w * v.w;
    #pragma unroll
    for (int m = 1; m < 32; m <<= 1) s += __shfl_xor(s, m, 32);
    float inv = 1.0f / fmaxf(sqrtf(s), CEPS);   // reference: z / max(||z||, eps)
    v4h h;
    h.x = (_Float16)(v.x * inv);
    h.y = (_Float16)(v.y * inv);
    h.z = (_Float16)(v.z * inv);
    h.w = (_Float16)(v.w * inv);
    *(v4h*)(zn + (size_t)row * NDIM + lane * 4) = h;
}

// ---------------------------------------------------------------------------
// Kernel 2: one block per 16-row stripe; 8 waves; wave w sweeps column tiles
// w, w+8, ... (64 tiles), ping-ponging two B-tile register buffers.
//
// Register images follow the CDNA5 ISA tables:
//   A (16x32 f16): lane<16 -> row=lane,   K chunks {0..7},{16..23} (+8 shift
//                  for lanes 16..31) => two 16B loads at (half?8:0), +16.
//   B (32x16 f16): lane<16 -> col=lane, K chunks {0..7},{16..23} as A (+8 for 16..31)
//                  => two 16B loads at (half?8:0), +16.
//   C/D (16x16 f32): VGPR r -> row = r + (half?8:0), col = lane%16.
// ---------------------------------------------------------------------------
__global__ __launch_bounds__(256)
void nxt_simloss_kernel(const _Float16* __restrict__ zn,
                        const int* __restrict__ labels,
                        float* __restrict__ partials) {
    __shared__ float sp[8][16];
    __shared__ float st[8][16];
    __shared__ float sl[16];

    const int lane  = threadIdx.x & 31;
    const int wave  = threadIdx.x >> 5;
    const int half  = (lane < 16) ? 0 : 1;
    const int l16   = lane & 15;
    const int ibase = blockIdx.x * 16;

    // A stripe (16 rows x 128 K) held in 4 x v16h for the whole sweep.
    v16h a[4];
    const _Float16* arow = zn + (size_t)(ibase + l16) * NDIM;
    #pragma unroll
    for (int kk = 0; kk < 4; ++kk) {
        const int ka = kk * 32 + half * 8;
        HalfPack u;
        u.q[0] = *(const uint4*)(arow + ka);        // K = ka .. ka+7
        u.q[1] = *(const uint4*)(arow + ka + 16);   // K = ka+16 .. ka+23
        a[kk] = u.h;
    }

    int labi[8];
    #pragma unroll
    for (int r = 0; r < 8; ++r) labi[r] = labels[ibase + r + half * 8];

    float pos[8], tot[8];
    #pragma unroll
    for (int r = 0; r < 8; ++r) { pos[r] = 0.0f; tot[r] = 0.0f; }

    auto loadbt = [&](BTile& bt, int t) {
        const int jbase = t * 16;
        const _Float16* brow = zn + (size_t)(jbase + l16) * NDIM;
        #pragma unroll
        for (int kk = 0; kk < 4; ++kk) {
            const int kb = kk * 32 + half * 8;
            HalfPack u;
            u.q[0] = *(const uint4*)(brow + kb);       // K = kb .. kb+7
            u.q[1] = *(const uint4*)(brow + kb + 16);  // K = kb+16 .. kb+23
            bt.b[kk] = u.h;
        }
        bt.labj = labels[jbase + l16];
    };

    auto compute = [&](const BTile& bt, int t) {
        v8f c = {};
        #pragma unroll
        for (int kk = 0; kk < 4; ++kk)
            c = __builtin_amdgcn_wmma_f32_16x16x32_f16(
                    false, a[kk], false, bt.b[kk], (short)0, c, false, false);
        const int j = t * 16 + l16;
        #pragma unroll
        for (int r = 0; r < 8; ++r) {
            const int i = ibase + r + half * 8;
            const float val = c[r];
            tot[r] += val;                                         // diag in tot
            pos[r] += (labi[r] == bt.labj && i != j) ? val : 0.0f; // diag excluded
        }
    };

    // Software-pipelined sweep: 64 tiles per wave = 32 ping-pong pairs.
    BTile b0, b1;
    int t0 = wave;
    loadbt(b0, t0);
    for (int it = 0; it < 31; ++it) {
        loadbt(b1, t0 + 8);       // prefetch while b0's WMMAs run
        compute(b0, t0);
        loadbt(b0, t0 + 16);      // prefetch while b1's WMMAs run
        compute(b1, t0 + 8);
        t0 += 16;
    }
    loadbt(b1, t0 + 8);
    compute(b0, t0);
    compute(b1, t0 + 8);

    // Reduce across the 16 lanes of each half-wave (xor<16 keeps the half).
    #pragma unroll
    for (int m = 1; m < 16; m <<= 1) {
        #pragma unroll
        for (int r = 0; r < 8; ++r) {
            pos[r] += __shfl_xor(pos[r], m, 32);
            tot[r] += __shfl_xor(tot[r], m, 32);
        }
    }
    if (l16 == 0) {   // lane 0 -> rows 0..7, lane 16 -> rows 8..15
        #pragma unroll
        for (int r = 0; r < 8; ++r) {
            sp[wave][r + half * 8] = pos[r];
            st[wave][r + half * 8] = tot[r];
        }
    }
    __syncthreads();

    // Fixed-order cross-wave reduction (deterministic, no float atomics).
    if (threadIdx.x < 16) {
        float p = 0.0f, q = 0.0f;
        #pragma unroll
        for (int w = 0; w < 8; ++w) { p += sp[w][threadIdx.x]; q += st[w][threadIdx.x]; }
        sl[threadIdx.x] = -logf(p / q);
    }
    __syncthreads();
    if (threadIdx.x == 0) {
        float s = 0.0f;
        #pragma unroll
        for (int m = 0; m < 16; ++m) s += sl[m];
        partials[blockIdx.x] = s;
    }
}

// ---------------------------------------------------------------------------
// Kernel 3: deterministic tree reduction of 512 partials -> mean loss.
// ---------------------------------------------------------------------------
__global__ __launch_bounds__(256)
void nxt_reduce_kernel(const float* __restrict__ partials, float* __restrict__ out) {
    __shared__ float s[256];
    s[threadIdx.x] = partials[threadIdx.x] + partials[threadIdx.x + 256];
    __syncthreads();
    #pragma unroll
    for (int m = 128; m > 0; m >>= 1) {
        if (threadIdx.x < m) s[threadIdx.x] += s[threadIdx.x + m];
        __syncthreads();
    }
    if (threadIdx.x == 0) out[0] = s[0] / (float)NROWS;
}

extern "C" void kernel_launch(void* const* d_in, const int* in_sizes, int n_in,
                              void* d_out, int out_size, void* d_ws, size_t ws_size,
                              hipStream_t stream) {
    const float* z      = (const float*)d_in[0];
    const int*   labels = (const int*)d_in[1];

    _Float16* zn       = (_Float16*)d_ws;                                   // 2 MB
    float*    partials = (float*)((char*)d_ws + (size_t)NROWS * NDIM * 2);  // 2 KB

    nxt_normalize_kernel<<<NROWS / 8, 256, 0, stream>>>(z, zn);
    nxt_simloss_kernel<<<NTILE, 256, 0, stream>>>(zn, labels, partials);
    nxt_reduce_kernel<<<1, 256, 0, stream>>>(partials, (float*)d_out);
}
